// Encoder_6588479832687
// MI455X (gfx1250) — compile-verified, matrix operand order checked by reading, not run on hardware
//
#include <hip/hip_runtime.h>

#define T_STEPS 256
#define BATCH   64
#define INP     512
#define HID     1024
#define NGATES  4096   // 4*HID
#define NLAYERS 4
#define KSPLIT  2
#define NBLOCKS (64 * KSPLIT)   // 64 row-slices x 2 K-slices

typedef unsigned short ushort_t;
typedef __attribute__((ext_vector_type(16))) __bf16 v16bf;
typedef __attribute__((ext_vector_type(8)))  float  v8f;

union FragU { v16bf v; uint4 q[2]; };

__device__ __forceinline__ ushort_t f2bf(float x) {
  unsigned u = __float_as_uint(x);
  u = (u + 0x7FFFu + ((u >> 16) & 1u)) >> 16;   // round-to-nearest-even
  return (ushort_t)u;
}

__global__ void cvt_bf16_kernel(const float* __restrict__ s, ushort_t* __restrict__ d, int n) {
  int i = blockIdx.x * blockDim.x + threadIdx.x;
  if (i < n) d[i] = f2bf(s[i]);
}

__global__ void vec_add_kernel(const float* __restrict__ a, const float* __restrict__ b,
                               float* __restrict__ d, int n) {
  int i = blockIdx.x * blockDim.x + threadIdx.x;
  if (i < n) d[i] = a[i] + b[i];
}

// Fragment from LDS tile stored row-major [rows][64] bf16, k-subchunk kc (0/1).
// 16-bit WMMA operand layout: lanes<16 hold k = {0..7, 16..23}, lanes>=16 hold
// k = {8..15, 24..31} -> two b128 reads per lane.
__device__ __forceinline__ v16bf lds_frag64(const ushort_t* base, int row, int lanehi, int kc) {
  FragU f;
  const uint4* p = (const uint4*)(base + row * 64 + kc * 32 + lanehi * 8);
  f.q[0] = p[0];
  f.q[1] = p[2];
  return f.v;
}

__device__ __forceinline__ void grid_sync(unsigned* cnt, unsigned* gen) {
  __threadfence();          // make this thread's stores device-visible
  __syncthreads();
  if (threadIdx.x == 0) {
    unsigned g = __hip_atomic_load(gen, __ATOMIC_ACQUIRE, __HIP_MEMORY_SCOPE_AGENT);
    unsigned arrived = __hip_atomic_fetch_add(cnt, 1u, __ATOMIC_ACQ_REL, __HIP_MEMORY_SCOPE_AGENT);
    if (arrived == NBLOCKS - 1u) {
      __hip_atomic_store(cnt, 0u, __ATOMIC_RELAXED, __HIP_MEMORY_SCOPE_AGENT);
      __hip_atomic_fetch_add(gen, 1u, __ATOMIC_RELEASE, __HIP_MEMORY_SCOPE_AGENT);
    } else {
      while (__hip_atomic_load(gen, __ATOMIC_ACQUIRE, __HIP_MEMORY_SCOPE_AGENT) == g) {
        __builtin_amdgcn_s_sleep(2);
      }
    }
  }
  __syncthreads();
}

// One persistent kernel runs the whole 255-step, 4-layer recurrence.
// Per (t,l): phase A = gatesPart[ks][4096][64] partial GEMMs (WMMA bf16, K split
//            across 2 block slices); phase B = sum partials + LSTM cell pointwise.
__global__ __launch_bounds__(256) void lstm_persistent(
    const ushort_t* __restrict__ wih0, const ushort_t* __restrict__ whh0,
    const ushort_t* __restrict__ wihr, const ushort_t* __restrict__ whhr,
    const ushort_t* __restrict__ xbf,  ushort_t* __restrict__ hbf,
    const float* __restrict__ bias, float* __restrict__ cT,
    float* __restrict__ gatesT, float* __restrict__ out,
    unsigned* __restrict__ bar_cnt, unsigned* __restrict__ bar_gen)
{
  __shared__ ushort_t sA[2][64 * 64];   // 8KB x2 : 64 gate rows x 64 k
  __shared__ ushort_t sB[2][64 * 64];   // 8KB x2 : 64 batch rows x 64 k

  const int tid    = threadIdx.x;
  const int wave   = tid >> 5;          // 0..7
  const int lane   = tid & 31;
  const int row16  = lane & 15;
  const int lanehi = lane >> 4;
  const int mgroup = wave >> 2;         // 0..1 -> 32-row half of the block tile
  const int ntile  = wave & 3;          // 0..3 -> 16 batch columns
  const int rslice = blockIdx.x & 63;   // row slice
  const int kslice = blockIdx.x >> 6;   // K slice (0/1)
  const int mbase  = rslice * 64;
  const int mloc   = lanehi * 8;
  const int ncol   = ntile * 16 + row16;
  float* gpart = gatesT + (size_t)kslice * NGATES * BATCH;

  for (int t = 1; t < T_STEPS; ++t) {
    const int wo = (t == T_STEPS - 1);
    for (int l = 0; l < NLAYERS; ++l) {
      const ushort_t* Wx = (l == 0) ? wih0 : wihr + (size_t)(l - 1) * NGATES * HID;
      const ushort_t* Wh = (l == 0) ? whh0 : whhr + (size_t)(l - 1) * NGATES * HID;
      const int Kx       = (l == 0) ? INP : HID;
      const ushort_t* X  = (l == 0) ? xbf + (size_t)t * INP
                                    : hbf + (size_t)(l - 1) * BATCH * HID;
      const int xstride  = (l == 0) ? T_STEPS * INP : HID;
      const ushort_t* Hb = hbf + (size_t)l * BATCH * HID;

      // ---------------- phase A: partial gate GEMM over this block's K slice ----
      v8f acc0 = {0.f,0.f,0.f,0.f,0.f,0.f,0.f,0.f};
      v8f acc1 = {0.f,0.f,0.f,0.f,0.f,0.f,0.f,0.f};

      const int nchx = Kx / 64;                 // x-part chunks (8 or 16)
      const int ncht = nchx + HID / 64;         // total chunks (24 or 32)
      const int half = ncht >> 1;
      const int c0 = kslice * half;
      const int c1 = c0 + half;

      // virtualized chunk c -> operand pointers
      auto stage = [&](int buf, int c) {
        const ushort_t* W; const ushort_t* Xp; int K, xs, kb;
        if (c < nchx) { W = Wx; Xp = X;  K = Kx;  xs = xstride; kb = c * 64; }
        else          { W = Wh; Xp = Hb; K = HID; xs = HID;     kb = (c - nchx) * 64; }
        #pragma unroll
        for (int i = 0; i < 2; ++i) {
          const int idx = tid + i * 256;         // 0..511
          const int r   = idx >> 3;              // row 0..63
          const int c8  = (idx & 7) * 8;         // k offset 0..56
          *(uint4*)(&sA[buf][r * 64 + c8]) =
              *(const uint4*)(W + (size_t)(mbase + r) * K + kb + c8);
          *(uint4*)(&sB[buf][r * 64 + c8]) =
              *(const uint4*)(Xp + (size_t)r * xs + kb + c8);
        }
      };
      auto compute = [&](int buf) {
        #pragma unroll
        for (int kc = 0; kc < 2; ++kc) {
          v16bf bf = lds_frag64(sB[buf], ntile * 16 + row16, lanehi, kc);
          v16bf a0 = lds_frag64(sA[buf], mgroup * 32 + row16, lanehi, kc);
          v16bf a1 = lds_frag64(sA[buf], mgroup * 32 + 16 + row16, lanehi, kc);
          acc0 = __builtin_amdgcn_wmma_f32_16x16x32_bf16(false, a0, false, bf, (short)0, acc0, false, false);
          acc1 = __builtin_amdgcn_wmma_f32_16x16x32_bf16(false, a1, false, bf, (short)0, acc1, false, false);
        }
      };

      stage(0, c0);
      __syncthreads();
      for (int ch = c0; ch < c1; ++ch) {
        const int cur = (ch - c0) & 1;
        if (ch + 1 < c1) stage(cur ^ 1, ch + 1);
        compute(cur);
        __syncthreads();   // one barrier per 4 WMMAs per wave
      }

      // epilogue: D layout -> partial gatesT rows = gate units, cols = batch
      #pragma unroll
      for (int r = 0; r < 8; ++r) {
        gpart[(size_t)(mbase + mgroup * 32 +      mloc + r) * BATCH + ncol] = acc0[r];
        gpart[(size_t)(mbase + mgroup * 32 + 16 + mloc + r) * BATCH + ncol] = acc1[r];
      }
      grid_sync(bar_cnt, bar_gen);

      // ---------------- phase B: sum K-partials + LSTM cell pointwise ----------
      {
        const float* g0 = gatesT;
        const float* g1 = gatesT + (size_t)NGATES * BATCH;
        const float* bsl = bias + (size_t)l * NGATES;
        float* cTl = cT + (size_t)l * HID * BATCH;
        ushort_t* hl = hbf + (size_t)l * BATCH * HID;
        float* oh = out + (size_t)l * BATCH * HID;
        float* oc = out + (size_t)(NLAYERS + l) * BATCH * HID;
        const int gt = blockIdx.x * 256 + tid;   // 0..32767
        #pragma unroll
        for (int e = 0; e < 2; ++e) {
          const int idx = gt + e * 32768;        // 0..65535 ; idx = n*64 + b
          const int n = idx >> 6;
          const int b = idx & 63;
          float gi = g0[idx]                   + g1[idx]                   + bsl[n];
          float gf = g0[1 * HID * BATCH + idx] + g1[1 * HID * BATCH + idx] + bsl[1 * HID + n];
          float gg = g0[2 * HID * BATCH + idx] + g1[2 * HID * BATCH + idx] + bsl[2 * HID + n];
          float go = g0[3 * HID * BATCH + idx] + g1[3 * HID * BATCH + idx] + bsl[3 * HID + n];
          gi = 1.f / (1.f + __expf(-gi));
          gf = 1.f / (1.f + __expf(-gf));
          gg = tanhf(gg);
          go = 1.f / (1.f + __expf(-go));
          float c = gf * cTl[idx] + gi * gg;
          float h = go * tanhf(c);
          cTl[idx] = c;
          hl[b * HID + n] = f2bf(h);
          if (wo) { oh[b * HID + n] = h; oc[b * HID + n] = c; }
        }
      }
      grid_sync(bar_cnt, bar_gen);
    }
  }
}

extern "C" void kernel_launch(void* const* d_in, const int* in_sizes, int n_in,
                              void* d_out, int out_size, void* d_ws, size_t ws_size,
                              hipStream_t stream) {
  const float* batch     = (const float*)d_in[0];
  const float* W_ih0     = (const float*)d_in[1];
  const float* W_hh0     = (const float*)d_in[2];
  const float* b_ih0     = (const float*)d_in[3];
  const float* b_hh0     = (const float*)d_in[4];
  const float* W_ih_rest = (const float*)d_in[5];
  const float* W_hh_rest = (const float*)d_in[6];
  const float* b_ih_rest = (const float*)d_in[7];
  const float* b_hh_rest = (const float*)d_in[8];
  float* out = (float*)d_out;

  // workspace carve-up (~83 MB)
  char* w = (char*)d_ws;
  ushort_t* wih0 = (ushort_t*)w;  w += sizeof(ushort_t) * (size_t)NGATES * INP;
  ushort_t* whh0 = (ushort_t*)w;  w += sizeof(ushort_t) * (size_t)NGATES * HID;
  ushort_t* wihr = (ushort_t*)w;  w += sizeof(ushort_t) * (size_t)3 * NGATES * HID;
  ushort_t* whhr = (ushort_t*)w;  w += sizeof(ushort_t) * (size_t)3 * NGATES * HID;
  ushort_t* xbf  = (ushort_t*)w;  w += sizeof(ushort_t) * (size_t)BATCH * T_STEPS * INP;
  ushort_t* hbf  = (ushort_t*)w;  w += sizeof(ushort_t) * (size_t)NLAYERS * BATCH * HID;
  float* bias    = (float*)w;     w += sizeof(float) * (size_t)NLAYERS * NGATES;
  float* cT      = (float*)w;     w += sizeof(float) * (size_t)NLAYERS * HID * BATCH;
  float* gatesT  = (float*)w;     w += sizeof(float) * (size_t)KSPLIT * NGATES * BATCH;
  unsigned* bars = (unsigned*)w;  w += 256;   // bar_cnt at [0], bar_gen at [1]

  // one-time prep: bf16 weights/input, fused biases, zeroed state
  {
    int n;
    n = NGATES * INP;           cvt_bf16_kernel<<<(n + 255) / 256, 256, 0, stream>>>(W_ih0, wih0, n);
    n = NGATES * HID;           cvt_bf16_kernel<<<(n + 255) / 256, 256, 0, stream>>>(W_hh0, whh0, n);
    n = 3 * NGATES * HID;       cvt_bf16_kernel<<<(n + 255) / 256, 256, 0, stream>>>(W_ih_rest, wihr, n);
    n = 3 * NGATES * HID;       cvt_bf16_kernel<<<(n + 255) / 256, 256, 0, stream>>>(W_hh_rest, whhr, n);
    n = BATCH * T_STEPS * INP;  cvt_bf16_kernel<<<(n + 255) / 256, 256, 0, stream>>>(batch, xbf, n);
  }
  vec_add_kernel<<<(NGATES + 255) / 256, 256, 0, stream>>>(b_ih0, b_hh0, bias, NGATES);
  for (int l = 1; l < NLAYERS; ++l)
    vec_add_kernel<<<(NGATES + 255) / 256, 256, 0, stream>>>(
        b_ih_rest + (size_t)(l - 1) * NGATES, b_hh_rest + (size_t)(l - 1) * NGATES,
        bias + (size_t)l * NGATES, NGATES);

  hipMemsetAsync(hbf, 0, sizeof(ushort_t) * (size_t)NLAYERS * BATCH * HID, stream);
  hipMemsetAsync(cT,  0, sizeof(float)    * (size_t)NLAYERS * HID * BATCH, stream);
  hipMemsetAsync(bars, 0, 256, stream);

  // whole recurrence in one persistent kernel (128 blocks, grid-sync between phases)
  lstm_persistent<<<NBLOCKS, 256, 0, stream>>>(
      wih0, whh0, wihr, whhr, xbf, hbf, bias, cT, gatesT, out,
      bars + 0, bars + 1);
}
